// WinogradConv2d_45088566673491
// MI455X (gfx1250) — matrix operand order repaired and checked by reading, not compiled, not run on hardware
//
#include <hip/hip_runtime.h>

// ---------------------------------------------------------------------------
// Winograd F(2x2,3x3) conv for MI455X (gfx1250, wave32, WMMA).
// GEMM stage runs as 16 per-frequency GEMMs on v_wmma_f32_16x16x32_bf16
// (fp32 accumulate); all transforms are fp32.
// ---------------------------------------------------------------------------

typedef __attribute__((ext_vector_type(16))) __bf16 v16bf;
typedef __attribute__((ext_vector_type(8)))  __bf16 v8bf;
typedef __attribute__((ext_vector_type(8)))  float  v8f;
typedef __attribute__((ext_vector_type(2)))  float  f32x2;

#define B_    8
#define C_    64
#define K_    64
#define H_    130
#define W_    130
#define NH_   64
#define NW_   64
#define HOUT_ 128
#define WOUT_ 128
#define TILES_PER_WG 16
#define VSTRIDE 72   // padded channel stride (bf16 elems): 144B = 36 dwords -> conflict-free

// ---------------------------------------------------------------------------
// Kernel 1: weight transform  U[xy][k][c] = (G w G^T), stored bf16 in d_ws.
// Layout [xy][k][c] makes the WMMA B-fragment two contiguous 16B loads.
// ---------------------------------------------------------------------------
__global__ __launch_bounds__(256)
void winograd_weight_xform(const float* __restrict__ w, __bf16* __restrict__ Ug) {
  int kc = blockIdx.x * blockDim.x + threadIdx.x;   // 4096 = K*C
  if (kc >= K_ * C_) return;
  int k = kc >> 6;
  int c = kc & 63;
  float g[3][3];
  #pragma unroll
  for (int i = 0; i < 3; ++i)
    #pragma unroll
    for (int j = 0; j < 3; ++j)
      g[i][j] = w[kc * 9 + i * 3 + j];
  float t[4][3];
  #pragma unroll
  for (int j = 0; j < 3; ++j) {
    t[0][j] = g[0][j];
    t[1][j] = 0.5f * (g[0][j] + g[1][j] + g[2][j]);
    t[2][j] = 0.5f * (g[0][j] - g[1][j] + g[2][j]);
    t[3][j] = g[2][j];
  }
  #pragma unroll
  for (int i = 0; i < 4; ++i) {
    float u0 = t[i][0];
    float u1 = 0.5f * (t[i][0] + t[i][1] + t[i][2]);
    float u2 = 0.5f * (t[i][0] - t[i][1] + t[i][2]);
    float u3 = t[i][2];
    Ug[(((i * 4 + 0) * K_ + k) * C_) + c] = (__bf16)u0;
    Ug[(((i * 4 + 1) * K_ + k) * C_) + c] = (__bf16)u1;
    Ug[(((i * 4 + 2) * K_ + k) * C_) + c] = (__bf16)u2;
    Ug[(((i * 4 + 3) * K_ + k) * C_) + c] = (__bf16)u3;
  }
}

// ---------------------------------------------------------------------------
// Kernel 2: per WG, 16 Winograd tiles x all 64 in/out channels.
//   Stage A: fp32 input transform B^T d B -> bf16 LDS V[xy][tile][c].
//   Stage B: 4 waves, each one 16-wide N-block; 16 freq x (Kdim 64 = 2 wmma).
//   Output transform A^T M A is per-lane fp32 (C/D layout keeps all 16 freq
//   values of one (tile,k) pair in the same lane).
// ---------------------------------------------------------------------------
__global__ __launch_bounds__(128)
void winograd_main(const float* __restrict__ x, const __bf16* __restrict__ Ug,
                   const float* __restrict__ bias, float* __restrict__ out) {
  __shared__ __align__(16) __bf16 Vlds[16 * TILES_PER_WG * VSTRIDE]; // 36 KB

  const int tid   = threadIdx.x;
  const int tile0 = blockIdx.x * TILES_PER_WG;

  // ---- Stage A: load patches, transform, scatter bf16 to LDS ----
  #pragma unroll
  for (int r = 0; r < 8; ++r) {
    const int tl = tid & 15;              // tile within WG (coalesced along m)
    const int c  = r * 8 + (tid >> 4);    // channel 0..63
    const int t  = tile0 + tl;
    const int mi = t & (NW_ - 1);
    const int ni = (t >> 6) & (NH_ - 1);
    const int b  = t >> 12;
    const float* p = x + (((b * C_ + c) * H_) + 2 * ni) * W_ + 2 * mi;
    float d[4][4];
    #pragma unroll
    for (int i = 0; i < 4; ++i) {
      f32x2 lo = *(const f32x2*)(p + i * W_);
      f32x2 hi = *(const f32x2*)(p + i * W_ + 2);
      d[i][0] = lo.x; d[i][1] = lo.y; d[i][2] = hi.x; d[i][3] = hi.y;
    }
    float z[4][4];
    #pragma unroll
    for (int j = 0; j < 4; ++j) {
      z[0][j] = d[0][j] - d[2][j];
      z[1][j] = d[1][j] + d[2][j];
      z[2][j] = d[2][j] - d[1][j];
      z[3][j] = d[1][j] - d[3][j];
    }
    #pragma unroll
    for (int i = 0; i < 4; ++i) {
      float v0 = z[i][0] - z[i][2];
      float v1 = z[i][1] + z[i][2];
      float v2 = z[i][2] - z[i][1];
      float v3 = z[i][1] - z[i][3];
      __bf16* vp = &Vlds[((i * 4) * TILES_PER_WG + tl) * VSTRIDE + c];
      vp[0 * TILES_PER_WG * VSTRIDE] = (__bf16)v0;   // xy = 4i+0
      vp[1 * TILES_PER_WG * VSTRIDE] = (__bf16)v1;   // xy = 4i+1
      vp[2 * TILES_PER_WG * VSTRIDE] = (__bf16)v2;   // xy = 4i+2
      vp[3 * TILES_PER_WG * VSTRIDE] = (__bf16)v3;   // xy = 4i+3
    }
  }
  __syncthreads();

  // ---- Stage B: WMMA GEMMs, one N-block per wave ----
  const int wave = tid >> 5;       // 0..3 -> which 16 output channels
  const int lane = tid & 31;
  const int half = lane >> 4;      // K-interleave half per ISA A/B layouts
  const int l15  = lane & 15;
  const int k    = wave * 16 + l15;

  v8f acc[16] = {};
  #pragma unroll
  for (int xy = 0; xy < 16; ++xy) {
    #pragma unroll
    for (int cc = 0; cc < 2; ++cc) {
      // A fragment (16x32 bf16): lane row = l15 (tile), per-ISA K interleave:
      //   elems 0..7  -> c = cc*32 + half*8 + i        (ds_load_b128)
      //   elems 8..15 -> c = cc*32 + 16 + half*8 + i   (ds_load_b128)
      const __bf16* ap = &Vlds[(xy * TILES_PER_WG + l15) * VSTRIDE + cc * 32 + half * 8];
      v8bf a0 = *(const v8bf*)ap;
      v8bf a1 = *(const v8bf*)(ap + 16);
      // B fragment (32x16 bf16): lane col = k; same K interleave as A
      // (c = cc*32 + half*8 + i, then +16) -> two global b128 loads.
      const __bf16* bp = Ug + ((xy * K_ + k) * C_) + cc * 32 + half * 8;
      v8bf b0 = *(const v8bf*)bp;
      v8bf b1 = *(const v8bf*)(bp + 16);
      v16bf Af, Bf;
      #pragma unroll
      for (int i = 0; i < 8; ++i) {
        Af[i] = a0[i]; Af[i + 8] = a1[i];
        Bf[i] = b0[i]; Bf[i + 8] = b1[i];
      }
      acc[xy] = __builtin_amdgcn_wmma_f32_16x16x32_bf16(
          /*neg_a=*/false, Af, /*neg_b=*/false, Bf,
          /*c_mod=*/(short)0, acc[xy], /*reuse_a=*/false, /*reuse_b=*/false);
    }
  }

  // ---- Output transform A^T M A + bias, per (lane, acc-VGPR) = (tile, k) ----
  const float bk = bias[k];
  #pragma unroll
  for (int v = 0; v < 8; ++v) {
    const int M  = v + 8 * half;          // C/D layout: row = vgpr + 8*(lane>=16)
    const int t  = tile0 + M;
    const int mi = t & (NW_ - 1);
    const int ni = (t >> 6) & (NH_ - 1);
    const int b  = t >> 12;
    float m0[4], m1[4], m2[4], m3[4];
    #pragma unroll
    for (int y = 0; y < 4; ++y) {
      m0[y] = acc[0 * 4 + y][v];
      m1[y] = acc[1 * 4 + y][v];
      m2[y] = acc[2 * 4 + y][v];
      m3[y] = acc[3 * 4 + y][v];
    }
    float t0[4], t1[4];
    #pragma unroll
    for (int y = 0; y < 4; ++y) {
      t0[y] = m0[y] + m1[y] + m2[y];
      t1[y] = m1[y] - m2[y] - m3[y];
    }
    f32x2 r0, r1;
    r0.x = t0[0] + t0[1] + t0[2] + bk;  r0.y = t0[1] - t0[2] - t0[3] + bk;
    r1.x = t1[0] + t1[1] + t1[2] + bk;  r1.y = t1[1] - t1[2] - t1[3] + bk;
    float* op = out + (((b * K_ + k) * HOUT_) + 2 * ni) * WOUT_ + 2 * mi;
    *(f32x2*)op = r0;
    *(f32x2*)(op + WOUT_) = r1;
  }
}

// ---------------------------------------------------------------------------
extern "C" void kernel_launch(void* const* d_in, const int* in_sizes, int n_in,
                              void* d_out, int out_size, void* d_ws, size_t ws_size,
                              hipStream_t stream) {
  (void)in_sizes; (void)n_in; (void)out_size; (void)ws_size;
  const float* x    = (const float*)d_in[0];
  const float* w    = (const float*)d_in[1];
  const float* bias = (const float*)d_in[2];
  float* out        = (float*)d_out;
  __bf16* Ug        = (__bf16*)d_ws;    // 16*64*64 bf16 = 128 KB scratch

  winograd_weight_xform<<<dim3((K_ * C_ + 255) / 256), dim3(256), 0, stream>>>(w, Ug);

  const int n_tiles = B_ * NH_ * NW_;   // 32768
  winograd_main<<<dim3(n_tiles / TILES_PER_WG), dim3(128), 0, stream>>>(x, Ug, bias, out);
}
